// SelfAttention_16423954939904
// MI455X (gfx1250) — matrix operand order repaired and checked by reading, not compiled, not run on hardware
//
#include <hip/hip_runtime.h>
#include <hip/hip_bf16.h>

typedef __attribute__((ext_vector_type(16))) _Float16 v16h;
typedef __attribute__((ext_vector_type(8)))  _Float16 v8h;
typedef __attribute__((ext_vector_type(8)))  float    v8f;
typedef __attribute__((ext_vector_type(4)))  unsigned int u32x4;
typedef __attribute__((ext_vector_type(8)))  int      i32x8;
typedef __attribute__((ext_vector_type(4)))  int      i32x4;

#define NN      2048
#define DD      64
#define NHEADS  64
#define CH      128                      /* K rows per TDM chunk (16 KB f16) */
#define BETA    0.125f                   /* 1/sqrt(64) */
#define SCALE   0.18033688011112042f     /* beta * log2(e) */
#define LN2     0.6931471805599453f

// ---------------------------------------------------------------------------
// Zero the scalar output (harness poisons d_out; we accumulate atomically).
// ---------------------------------------------------------------------------
__global__ void sa_zero_out(float* out) {
    if (threadIdx.x == 0) out[0] = 0.0f;
}

// ---------------------------------------------------------------------------
// Phase 1: Q[h][q][z] = sum_d x[q][d] * W[h][z][d]   (f32 in -> f16 out)
// One wave computes a 16x64 output tile for one head via v_wmma_f32_16x16x32_f16.
// blockIdx = (q-chunk of 128 rows, head, 0=Wq->Q / 1=Wk->K); 256 thr = 8 waves.
// ---------------------------------------------------------------------------
__global__ __launch_bounds__(256) void sa_proj_kernel(
    const float* __restrict__ x,  const float* __restrict__ Wq,
    const float* __restrict__ Wk, _Float16* __restrict__ Qws,
    _Float16* __restrict__ Kws)
{
    const int lane = threadIdx.x & 31;
    const int wave = threadIdx.x >> 5;
    const int l16  = lane & 15;
    const int hi   = lane >> 4;
    const int h    = blockIdx.y;
    const int q0   = blockIdx.x * 128 + wave * 16;

    const float* W   = blockIdx.z ? Wk : Wq;
    _Float16*    dst = (blockIdx.z ? Kws : Qws) + (size_t)h * NN * DD;

    // A-layout (contiguous): lane<16 -> K = {0..15}; lane>=16 -> K = {16..31}.
    const float* xr    = x + (q0 + l16) * DD;
    const int    koffA = hi * 16;
    v16h a0, a1;
#pragma unroll
    for (int i = 0; i < 8; ++i) {
        a0[i]     = (_Float16)xr[koffA + i];
        a0[8 + i] = (_Float16)xr[koffA + 8 + i];
        a1[i]     = (_Float16)xr[32 + koffA + i];
        a1[8 + i] = (_Float16)xr[32 + koffA + 8 + i];
    }

    const float* Wh    = W + (size_t)h * DD * DD;   // [z][d], d contiguous
    const int    koffB = hi * 16;
#pragma unroll
    for (int z0 = 0; z0 < DD; z0 += 16) {
        const float* wc = Wh + (z0 + l16) * DD;
        v16h b0, b1;
#pragma unroll
        for (int i = 0; i < 16; ++i) {
            b0[i] = (_Float16)wc[koffB + i];
            b1[i] = (_Float16)wc[32 + koffB + i];
        }
        v8f c = {};
        c = __builtin_amdgcn_wmma_f32_16x16x32_f16(false, a0, false, b0,
                                                   (short)0, c, false, false);
        c = __builtin_amdgcn_wmma_f32_16x16x32_f16(false, a1, false, b1,
                                                   (short)0, c, false, false);
#pragma unroll
        for (int v = 0; v < 8; ++v) {
            int m = v + hi * 8;
            dst[(size_t)(q0 + m) * DD + z0 + l16] = (_Float16)c[v];
        }
    }
}

// ---------------------------------------------------------------------------
// Phase 2: block = 8 waves, one head; wave owns 32 q-rows. K is staged through
// LDS by the Tensor Data Mover (wave 0 issues tensor_load_to_lds, double
// buffered, s_wait_tensorcnt + barrier). Per 32-column step: 8 WMMAs, then a
// per-lane online LSE update in the log2 domain (3 v_exp per 2 elements).
// Cross-lane merge happens once at the end.
// ---------------------------------------------------------------------------
__global__ __launch_bounds__(256) void sa_lse_kernel(
    const _Float16* __restrict__ Qws, const _Float16* __restrict__ Kws,
    float* __restrict__ out)
{
    // Only shared variable in this kernel -> group-segment offset 0 (the TDM
    // descriptor's lds_addr below relies on this).
    __shared__ __align__(16) _Float16 kbuf[2][CH * DD];   // 2 x 16 KB

    const int lane = threadIdx.x & 31;
    const int wave = threadIdx.x >> 5;
    const int l16  = lane & 15;
    const int hi   = lane >> 4;
    const int h    = blockIdx.y;
    const int q0   = blockIdx.x * 256 + wave * 32;   // 32 rows per wave

    const _Float16* Qh = Qws + (size_t)h * NN * DD;
    const _Float16* Kh = Kws + (size_t)h * NN * DD;

    // TDM: load K chunk (CH x 64 f16 tile of the 2048 x 64 tensor) into LDS.
    // D# packing per CDNA5 ISA §8.3-8.5 (group0: count=1, lds_addr, 57-bit
    // global addr, type=2; group1: data_size=2B, tensor_dim0=64,
    // tensor_dim1=2048, tile_dim0=64, tile_dim1=CH, dim0_stride=64).
    // This toolchain exposes the 6-arg builtin: groups 2/3 + trailing group
    // are unused for a 2-D tensor and passed as zeros.
    auto issue_tdm = [&](int chunk, int buf) {
        unsigned long long ga =
            (unsigned long long)(size_t)(Kh + (size_t)chunk * CH * DD);
        u32x4 g0 = {};
        i32x8 g1 = {};
        i32x4 g2 = {}, g3 = {};
        i32x8 g4 = {};
        g0[0] = 1u;                                       // count = 1
        g0[1] = buf ? (unsigned)(CH * DD * 2) : 0u;       // lds_addr (bytes)
        g0[2] = (unsigned)(ga & 0xffffffffull);           // global_addr[31:0]
        g0[3] = (unsigned)((ga >> 32) & 0x01ffffffull)    // global_addr[56:32]
              | 0x80000000u;                              // type = 2 (bits 127:126)
        g1[0] = (int)(1u << 16);                          // data_size = 1 (2 bytes)
        g1[1] = (int)(64u << 16);                         // tensor_dim0 = 64
        g1[2] = (int)((2048u & 0xffffu) << 16);           // tensor_dim1 = 2048
        g1[3] = (int)(64u << 16);                         // tile_dim0 = 64
        g1[4] = (int)CH;                                  // tile_dim1 = CH
        g1[5] = 64;                                       // tensor_dim0_stride = 64
        __builtin_amdgcn_tensor_load_to_lds(g0, g1, g2, g3, g4, 0);
    };

    // A fragments: rows [q0, q0+16) and [q0+16, q0+32), z halves 0/1.
    const int koffA = hi * 16;
    v16h aA0, aA1, aB0, aB1;
    {
        const _Float16* qa = Qh + (size_t)(q0 + l16) * DD;
        const _Float16* qb = Qh + (size_t)(q0 + 16 + l16) * DD;
        v8h t0 = *(const v8h*)(qa + koffA);
        v8h t1 = *(const v8h*)(qa + koffA + 8);
        v8h t2 = *(const v8h*)(qa + 32 + koffA);
        v8h t3 = *(const v8h*)(qa + 32 + koffA + 8);
        v8h u0 = *(const v8h*)(qb + koffA);
        v8h u1 = *(const v8h*)(qb + koffA + 8);
        v8h u2 = *(const v8h*)(qb + 32 + koffA);
        v8h u3 = *(const v8h*)(qb + 32 + koffA + 8);
#pragma unroll
        for (int i = 0; i < 8; ++i) {
            aA0[i] = t0[i]; aA0[8 + i] = t1[i];
            aA1[i] = t2[i]; aA1[8 + i] = t3[i];
            aB0[i] = u0[i]; aB0[8 + i] = u1[i];
            aB1[i] = u2[i]; aB1[8 + i] = u3[i];
        }
    }

    // Per-lane online LSE state in the log2 domain.
    float m[16], s[16];
#pragma unroll
    for (int v = 0; v < 16; ++v) { m[v] = -__builtin_inff(); s[v] = 0.0f; }

    // Preload chunk 0.
    if (wave == 0) { issue_tdm(0, 0); __builtin_amdgcn_s_wait_tensorcnt(0); }
    __syncthreads();

    const int koffB = hi * 16;
    const int NCHUNK = NN / CH;
    for (int c = 0; c < NCHUNK; ++c) {
        const _Float16* cur = kbuf[c & 1];
        if (wave == 0 && c + 1 < NCHUNK) issue_tdm(c + 1, (c + 1) & 1);

        // 32 K-columns per step: two B-fragment pairs feed 8 WMMAs.
#pragma unroll
        for (int kk = 0; kk < CH; kk += 32) {
            const _Float16* kr0 = cur + (kk + l16) * DD + koffB;
            const _Float16* kr1 = cur + (kk + 16 + l16) * DD + koffB;
            v8h t0 = *(const v8h*)(kr0);
            v8h t1 = *(const v8h*)(kr0 + 8);
            v8h t2 = *(const v8h*)(kr0 + 32);
            v8h t3 = *(const v8h*)(kr0 + 40);
            v8h u0 = *(const v8h*)(kr1);
            v8h u1 = *(const v8h*)(kr1 + 8);
            v8h u2 = *(const v8h*)(kr1 + 32);
            v8h u3 = *(const v8h*)(kr1 + 40);
            v16h b00, b01, b10, b11;
#pragma unroll
            for (int i = 0; i < 8; ++i) {
                b00[i] = t0[i]; b00[8 + i] = t1[i];
                b01[i] = t2[i]; b01[8 + i] = t3[i];
                b10[i] = u0[i]; b10[8 + i] = u1[i];
                b11[i] = u2[i]; b11[8 + i] = u3[i];
            }

            v8f cA0 = {}, cA1 = {}, cB0 = {}, cB1 = {};
            cA0 = __builtin_amdgcn_wmma_f32_16x16x32_f16(false, aA0, false, b00,
                                                         (short)0, cA0, false, false);
            cA0 = __builtin_amdgcn_wmma_f32_16x16x32_f16(false, aA1, false, b01,
                                                         (short)0, cA0, false, false);
            cA1 = __builtin_amdgcn_wmma_f32_16x16x32_f16(false, aA0, false, b10,
                                                         (short)0, cA1, false, false);
            cA1 = __builtin_amdgcn_wmma_f32_16x16x32_f16(false, aA1, false, b11,
                                                         (short)0, cA1, false, false);
            cB0 = __builtin_amdgcn_wmma_f32_16x16x32_f16(false, aB0, false, b00,
                                                         (short)0, cB0, false, false);
            cB0 = __builtin_amdgcn_wmma_f32_16x16x32_f16(false, aB1, false, b01,
                                                         (short)0, cB0, false, false);
            cB1 = __builtin_amdgcn_wmma_f32_16x16x32_f16(false, aB0, false, b10,
                                                         (short)0, cB1, false, false);
            cB1 = __builtin_amdgcn_wmma_f32_16x16x32_f16(false, aB1, false, b11,
                                                         (short)0, cB1, false, false);

            // Per-row online update, 2 new elements per row, 3 exp2 each.
#pragma unroll
            for (int v = 0; v < 8; ++v) {
                float va0 = cA0[v] * SCALE, va1 = cA1[v] * SCALE;
                float na  = fmaxf(m[v], fmaxf(va0, va1));
                s[v] = s[v] * exp2f(m[v] - na)
                     + exp2f(va0 - na) + exp2f(va1 - na);
                m[v] = na;

                float vb0 = cB0[v] * SCALE, vb1 = cB1[v] * SCALE;
                float nb  = fmaxf(m[8 + v], fmaxf(vb0, vb1));
                s[8 + v] = s[8 + v] * exp2f(m[8 + v] - nb)
                         + exp2f(vb0 - nb) + exp2f(vb1 - nb);
                m[8 + v] = nb;
            }
        }

        if (wave == 0) __builtin_amdgcn_s_wait_tensorcnt(0);
        __syncthreads();
    }

    // Merge (m, s) across the 16 lanes of each half-wave (log2-domain LSE
    // merge), then accumulate per-row natural-log LSEs.
    float total = 0.0f;
#pragma unroll
    for (int v = 0; v < 16; ++v) {
        float mm = m[v], ss = s[v];
#pragma unroll
        for (int off = 1; off < 16; off <<= 1) {
            float om = __shfl_xor(mm, off, 16);
            float os = __shfl_xor(ss, off, 16);
            float nm = fmaxf(mm, om);
            ss = ss * exp2f(mm - nm) + os * exp2f(om - nm);
            mm = nm;
        }
        total += LN2 * (mm + __log2f(ss));
    }
    // lane 0 carries rows {0..7, 16..23}, lane 16 carries rows {8..15, 24..31}.
    if (l16 == 0) atomicAdd(out, -8.0f * total);   // -1/beta = -8
}

// ---------------------------------------------------------------------------
extern "C" void kernel_launch(void* const* d_in, const int* in_sizes, int n_in,
                              void* d_out, int out_size, void* d_ws, size_t ws_size,
                              hipStream_t stream)
{
    (void)in_sizes; (void)n_in; (void)out_size; (void)ws_size;
    const float* x  = (const float*)d_in[0];
    const float* Wq = (const float*)d_in[1];
    const float* Wk = (const float*)d_in[2];
    float*       out = (float*)d_out;

    _Float16* Qws = (_Float16*)d_ws;                       // 64*2048*64 f16 = 16 MB
    _Float16* Kws = Qws + (size_t)NHEADS * NN * DD;        // + 16 MB

    sa_zero_out<<<1, 64, 0, stream>>>(out);
    sa_proj_kernel<<<dim3(NN / 128, NHEADS, 2), 256, 0, stream>>>(x, Wq, Wk, Qws, Kws);
    sa_lse_kernel<<<dim3(NN / 256, NHEADS), 256, 0, stream>>>(Qws, Kws, out);
}
